// TransformerBlock_18253611008342
// MI455X (gfx1250) — matrix operand order repaired and checked by reading, not compiled, not run on hardware
//
#include <hip/hip_runtime.h>
#include <hip/hip_bf16.h>

// ---------------------------------------------------------------------------
// CDNA5 (gfx1250) transformer block.
// bf16 WMMA GEMM core (wave32) with double-buffered LDS fed by
// GLOBAL_LOAD_ASYNC_TO_LDS_B128 (ASYNCcnt-tracked async copies).
// ---------------------------------------------------------------------------

typedef __bf16 bf16_t;
typedef __attribute__((ext_vector_type(16))) __bf16 v16bf;
typedef __attribute__((ext_vector_type(8)))  __bf16 v8bf;
typedef __attribute__((ext_vector_type(4)))  __bf16 v4bf;
typedef __attribute__((ext_vector_type(8)))  float  v8f;
typedef __attribute__((ext_vector_type(4)))  float  f32x4;

#define DIM_  2048
#define FFN_  8192
#define SEQ_  2048
#define MTOK_ 4096   // B*S

__device__ __forceinline__ float wave_sum(float v) {
  #pragma unroll
  for (int o = 16; o > 0; o >>= 1) v += __shfl_xor(v, o, 32);
  return v;
}
__device__ __forceinline__ float wave_max(float v) {
  #pragma unroll
  for (int o = 16; o > 0; o >>= 1) v = fmaxf(v, __shfl_xor(v, o, 32));
  return v;
}

// Generic LDS pointer -> LDS byte offset (flat aperture keeps the LDS byte
// address in the low 32 bits; ISA 10.2 aperture table).
__device__ __forceinline__ unsigned lds_off(const void* p) {
  return (unsigned)(unsigned long long)p;
}
// Per-lane async 16B copy global -> LDS.  ISA: GLOBAL_LOAD_ASYNC_TO_LDS_B128,
// tracked with ASYNCcnt (not LOADcnt/DScnt).
__device__ __forceinline__ void async_cp16(const void* gsrc, unsigned ldst) {
  asm volatile("global_load_async_to_lds_b128 %0, %1, off"
               :: "v"(ldst), "v"(gsrc) : "memory");
}
__device__ __forceinline__ void async_wait0() {
  asm volatile("s_wait_asynccnt 0x0" ::: "memory");
}

// ---------------------------------------------------------------------------
// fp32 -> bf16 bulk convert (weights / activations)
// ---------------------------------------------------------------------------
__global__ __launch_bounds__(256)
void cvt_bf16_kernel(const float* __restrict__ src, bf16_t* __restrict__ dst, int n)
{
  int i = (blockIdx.x * 256 + threadIdx.x) * 4;
  if (i < n) {
    f32x4 v = *(const f32x4*)(src + i);
    v4bf o;
    o.x = (bf16_t)v.x; o.y = (bf16_t)v.y; o.z = (bf16_t)v.z; o.w = (bf16_t)v.w;
    *(v4bf*)(dst + i) = o;
  }
}

// ---------------------------------------------------------------------------
// Additive-"rope" (optional) + RMSNorm.  One block (256 thr) per token.
// norm = scale * v / sqrt(sum(v^2) + eps)   (full L2, no mean)
// ---------------------------------------------------------------------------
template <bool ROPE>
__global__ __launch_bounds__(256)
void rmsnorm_kernel(const float* __restrict__ x, const float* __restrict__ scale,
                    float* __restrict__ xout, bf16_t* __restrict__ hbf)
{
  const int tok = blockIdx.x;
  const int s   = tok & (SEQ_ - 1);
  const float* xp = x + (size_t)tok * DIM_;
  float vals[8];
  float ss = 0.f;
  #pragma unroll
  for (int i = 0; i < 8; ++i) {
    int d = threadIdx.x + i * 256;
    float v = xp[d];
    if (ROPE) {
      int j = d & 1023;                       // table[s,d] = s * 10000^(-j/1024)
      float f = (float)s * exp2f(-(float)j * (13.287712379549449f / 1024.f));
      v += f;
      xout[(size_t)tok * DIM_ + d] = v;
    }
    vals[i] = v;
    ss += v * v;
  }
  __shared__ float red[8];
  ss = wave_sum(ss);
  if ((threadIdx.x & 31) == 0) red[threadIdx.x >> 5] = ss;
  __syncthreads();
  float tot = red[0] + red[1] + red[2] + red[3] + red[4] + red[5] + red[6] + red[7];
  float inv = rsqrtf(tot + 1e-5f);
  bf16_t* hp = hbf + (size_t)tok * DIM_;
  #pragma unroll
  for (int i = 0; i < 8; ++i) {
    int d = threadIdx.x + i * 256;
    hp[d] = (bf16_t)(scale[d] * vals[i] * inv);
  }
}

// ---------------------------------------------------------------------------
// bf16 WMMA GEMM:  C[M,N] = A[M,K] @ W[N,K]^T + bias (+resid), fp32 accum.
// Block: 256 threads (8 waves), 128(M) x 256(N) tile, K-step 32.
// Waves 2(M) x 4(N); each wave owns 64x64 = 4x4 WMMA tiles -> 16 wmma per
// K-step against 16 ds_load_b128 fragment loads (1:1).
// Global->LDS staging: double-buffered, GLOBAL_LOAD_ASYNC_TO_LDS_B128.
// Fragment layouts per ISA 7.12.2 (A: lane<16 K0-7/16-23, lane>=16 K8-15/24-31;
// B: col N=lane%16, same K split as A: lane<16 K0-7/16-23, lane>=16 K8-15/24-31).
// ---------------------------------------------------------------------------
template <bool OUT_BF16, bool ADD_RESID>
__global__ __launch_bounds__(256)
void gemm_bf16_wmma(const bf16_t* __restrict__ A,    // [M,K]
                    const bf16_t* __restrict__ Wt,   // [N,K] (torch W[out,in])
                    const float*  __restrict__ bias, // [N]
                    const float*  __restrict__ resid,// [M,N] fp32 or null
                    void*         __restrict__ Cout, // [M,N] f32 or bf16
                    int M, int N, int K)
{
  constexpr int BM = 128, BN = 256, BK = 32, LDK = BK + 8;  // 80B LDS rows
  constexpr int ATILE = BM * LDK;
  constexpr int BTILE = BN * LDK;
  constexpr int BUF   = ATILE + BTILE;
  __shared__ bf16_t smem[2 * BUF];                          // 60 KB

  const int tid  = threadIdx.x;
  const int lane = tid & 31;
  const int wid  = tid >> 5;
  const int m0   = blockIdx.y * BM;
  const int n0   = blockIdx.x * BN;
  const int wm   = (wid & 1) << 6;   // 0,64
  const int wn   = (wid >> 1) << 6;  // 0,64,128,192
  const int l16  = lane & 15;
  const int lhi  = lane >> 4;

  v8f acc[4][4] = {};

  // cooperative staging: 16B chunks; A rows crow,crow+64; B rows crow..crow+192
  const int crow = tid >> 2;         // 0..63
  const int ck8  = (tid & 3) << 3;   // bf16 elem offset 0,8,16,24

  auto issue_tile = [&](int kk, bf16_t* sbuf) {
    bf16_t* sAb = sbuf;
    bf16_t* sBb = sbuf + ATILE;
    const bf16_t* gA = A  + (size_t)(m0 + crow) * K + kk + ck8;
    const bf16_t* gB = Wt + (size_t)(n0 + crow) * K + kk + ck8;
    async_cp16(gA,                   lds_off(sAb + (size_t)(crow)       * LDK + ck8));
    async_cp16(gA + (size_t)64  * K, lds_off(sAb + (size_t)(crow +  64) * LDK + ck8));
    async_cp16(gB,                   lds_off(sBb + (size_t)(crow)       * LDK + ck8));
    async_cp16(gB + (size_t)64  * K, lds_off(sBb + (size_t)(crow +  64) * LDK + ck8));
    async_cp16(gB + (size_t)128 * K, lds_off(sBb + (size_t)(crow + 128) * LDK + ck8));
    async_cp16(gB + (size_t)192 * K, lds_off(sBb + (size_t)(crow + 192) * LDK + ck8));
  };

  auto compute_tile = [&](const bf16_t* sbuf) {
    const bf16_t* sAb = sbuf;
    const bf16_t* sBb = sbuf + ATILE;
    v16bf af[4];
    #pragma unroll
    for (int mt = 0; mt < 4; ++mt) {
      const bf16_t* base = sAb + (size_t)(wm + mt * 16 + l16) * LDK;
      v8bf lo = *(const v8bf*)(base + lhi * 8);
      v8bf hi = *(const v8bf*)(base + 16 + lhi * 8);
      af[mt] = __builtin_shufflevector(lo, hi, 0,1,2,3,4,5,6,7,8,9,10,11,12,13,14,15);
    }
    #pragma unroll
    for (int nt = 0; nt < 4; ++nt) {
      const bf16_t* base = sBb + (size_t)(wn + nt * 16 + l16) * LDK;
      v8bf lo = *(const v8bf*)(base + lhi * 8);
      v8bf hi = *(const v8bf*)(base + 16 + lhi * 8);
      v16bf bfr = __builtin_shufflevector(lo, hi, 0,1,2,3,4,5,6,7,8,9,10,11,12,13,14,15);
      #pragma unroll
      for (int mt = 0; mt < 4; ++mt) {
        acc[mt][nt] = __builtin_amdgcn_wmma_f32_16x16x32_bf16(
            false, af[mt], false, bfr, (short)0, acc[mt][nt], false, false);
      }
    }
  };

  // software pipeline: async-load k+1 while WMMAs chew on k
  issue_tile(0, smem);
  async_wait0();
  __syncthreads();
  int cur = 0;
  for (int kk = 0; kk < K; kk += BK) {
    if (kk + BK < K) issue_tile(kk + BK, smem + (size_t)(cur ^ 1) * BUF);
    compute_tile(smem + (size_t)cur * BUF);
    async_wait0();       // next tile fully in LDS
    __syncthreads();     // all waves done reading cur, next tile visible
    cur ^= 1;
  }

  // epilogue: D vgpr r holds M=r (lanes 0-15) / M=r+8 (lanes 16-31), N=lane%16
  #pragma unroll
  for (int nt = 0; nt < 4; ++nt) {
    const int col = n0 + wn + nt * 16 + l16;
    const float bv = bias[col];
    #pragma unroll
    for (int mt = 0; mt < 4; ++mt) {
      const int rbase = m0 + wm + mt * 16 + lhi * 8;
      #pragma unroll
      for (int r = 0; r < 8; ++r) {
        const size_t idx = (size_t)(rbase + r) * N + col;
        float v = acc[mt][nt][r] + bv;
        if (ADD_RESID) v += resid[idx];
        if (OUT_BF16) ((bf16_t*)Cout)[idx] = (bf16_t)v;
        else          ((float*)Cout)[idx]  = v;
      }
    }
  }
}

// ---------------------------------------------------------------------------
// Per-token "attention": scores[qh,kv] = q[qh]·k[kv]/8 over 32 heads, softmax
// over kv, ctx = attn @ v.  One 1024-thread block per token, 1 wave per qh.
// ---------------------------------------------------------------------------
__global__ __launch_bounds__(1024)
void attn_kernel(const float* __restrict__ qf, const float* __restrict__ kf,
                 const float* __restrict__ vf, bf16_t* __restrict__ ctx)
{
  __shared__ float sQ[32][65], sK[32][65], sV[32][65];
  const int tok = blockIdx.x;
  const size_t base = (size_t)tok * DIM_;
  for (int i = threadIdx.x; i < DIM_; i += 1024) {
    sQ[i >> 6][i & 63] = qf[base + i];
    sK[i >> 6][i & 63] = kf[base + i];
    sV[i >> 6][i & 63] = vf[base + i];
  }
  __syncthreads();
  const int qh   = threadIdx.x >> 5;
  const int lane = threadIdx.x & 31;
  float dot = 0.f;
  #pragma unroll
  for (int d = 0; d < 64; ++d) dot += sQ[qh][d] * sK[lane][d];
  float sc  = dot * 0.125f;                 // / sqrt(64)
  float mx  = wave_max(sc);
  float p   = __expf(sc - mx);
  float sum = wave_sum(p);
  float attn = p / sum;
  float acc0 = 0.f, acc1 = 0.f;
  #pragma unroll
  for (int kv = 0; kv < 32; ++kv) {
    float a = __shfl(attn, kv, 32);
    acc0 += a * sV[kv][lane];
    acc1 += a * sV[kv][lane + 32];
  }
  bf16_t* cp = ctx + base + qh * 64;
  cp[lane]      = (bf16_t)acc0;
  cp[lane + 32] = (bf16_t)acc1;
}

// ---------------------------------------------------------------------------
// out = xres + silu(g1) * g2
// ---------------------------------------------------------------------------
__global__ __launch_bounds__(256)
void swiglu_kernel(const float* __restrict__ xres, const float* __restrict__ g1,
                   const float* __restrict__ g2, float* __restrict__ out, int n)
{
  int i = blockIdx.x * 256 + threadIdx.x;
  if (i < n) {
    float z = g1[i];
    float s = z / (1.f + __expf(-z));
    out[i] = xres[i] + s * g2[i];
  }
}

// ---------------------------------------------------------------------------
extern "C" void kernel_launch(void* const* d_in, const int* in_sizes, int n_in,
                              void* d_out, int out_size, void* d_ws, size_t ws_size,
                              hipStream_t stream) {
  (void)in_sizes; (void)n_in; (void)out_size; (void)ws_size;
  const float* x      = (const float*)d_in[0];
  const float* wq     = (const float*)d_in[1];
  const float* bq     = (const float*)d_in[2];
  const float* wk     = (const float*)d_in[3];
  const float* bk     = (const float*)d_in[4];
  const float* wv     = (const float*)d_in[5];
  const float* bv     = (const float*)d_in[6];
  const float* wo     = (const float*)d_in[7];
  const float* bo     = (const float*)d_in[8];
  const float* scale1 = (const float*)d_in[9];
  const float* scale2 = (const float*)d_in[10];
  const float* w_in   = (const float*)d_in[11];
  const float* b_in   = (const float*)d_in[12];
  const float* w1     = (const float*)d_in[13];
  const float* b1     = (const float*)d_in[14];
  const float* w2     = (const float*)d_in[15];
  const float* b2     = (const float*)d_in[16];

  constexpr size_t MB = 1ull << 20;
  char* ws = (char*)d_ws;
  float*  x1    = (float*)(ws + 0);            // 32MB  x + rope
  float*  x2    = (float*)(ws + 32 * MB);      // 32MB  post-attn residual
  bf16_t* hbf   = (bf16_t*)(ws + 64 * MB);     // 16MB  normed activations
  bf16_t* ctxbf = (bf16_t*)(ws + 80 * MB);     // 16MB
  float*  qf    = (float*)(ws + 96 * MB);      // 32MB  (reused as g1)
  float*  kf    = (float*)(ws + 128 * MB);     // 32MB  (reused as g2)
  float*  vfb   = (float*)(ws + 160 * MB);     // 32MB
  bf16_t* hidbf = (bf16_t*)(ws + 192 * MB);    // 64MB
  bf16_t* wqb   = (bf16_t*)(ws + 256 * MB);    // 8MB each
  bf16_t* wkb   = (bf16_t*)(ws + 264 * MB);
  bf16_t* wvb   = (bf16_t*)(ws + 272 * MB);
  bf16_t* wob   = (bf16_t*)(ws + 280 * MB);
  bf16_t* winb  = (bf16_t*)(ws + 288 * MB);    // 32MB
  bf16_t* w1b   = (bf16_t*)(ws + 320 * MB);    // 32MB
  bf16_t* w2b   = (bf16_t*)(ws + 352 * MB);    // 32MB

  const int DD = DIM_ * DIM_, FD = FFN_ * DIM_;
  cvt_bf16_kernel<<<DD / 1024, 256, 0, stream>>>(wq, wqb, DD);
  cvt_bf16_kernel<<<DD / 1024, 256, 0, stream>>>(wk, wkb, DD);
  cvt_bf16_kernel<<<DD / 1024, 256, 0, stream>>>(wv, wvb, DD);
  cvt_bf16_kernel<<<DD / 1024, 256, 0, stream>>>(wo, wob, DD);
  cvt_bf16_kernel<<<FD / 1024, 256, 0, stream>>>(w_in, winb, FD);
  cvt_bf16_kernel<<<FD / 1024, 256, 0, stream>>>(w1, w1b, FD);
  cvt_bf16_kernel<<<FD / 1024, 256, 0, stream>>>(w2, w2b, FD);

  // x1 = x + rope ; hbf = bf16(rmsnorm(x1, scale1))
  rmsnorm_kernel<true><<<MTOK_, 256, 0, stream>>>(x, scale1, x1, hbf);

  dim3 gDD(DIM_ / 256, MTOK_ / 128);
  gemm_bf16_wmma<false, false><<<gDD, 256, 0, stream>>>(hbf, wqb, bq, nullptr, qf,  MTOK_, DIM_, DIM_);
  gemm_bf16_wmma<false, false><<<gDD, 256, 0, stream>>>(hbf, wkb, bk, nullptr, kf,  MTOK_, DIM_, DIM_);
  gemm_bf16_wmma<false, false><<<gDD, 256, 0, stream>>>(hbf, wvb, bv, nullptr, vfb, MTOK_, DIM_, DIM_);

  attn_kernel<<<MTOK_, 1024, 0, stream>>>(qf, kf, vfb, ctxbf);

  // x2 = x1 + ctx @ wo^T + bo
  gemm_bf16_wmma<false, true><<<gDD, 256, 0, stream>>>(ctxbf, wob, bo, x1, x2, MTOK_, DIM_, DIM_);

  // hbf = bf16(rmsnorm(x2, scale2))
  rmsnorm_kernel<false><<<MTOK_, 256, 0, stream>>>(x2, scale2, nullptr, hbf);

  // hidden(bf16) = h2 @ w_in^T + b_in
  dim3 gFD(FFN_ / 256, MTOK_ / 128);
  gemm_bf16_wmma<true, false><<<gFD, 256, 0, stream>>>(hbf, winb, b_in, nullptr, hidbf, MTOK_, FFN_, DIM_);

  // g1/g2 = hidden @ w1^T + b1, hidden @ w2^T + b2  (reuse qf/kf)
  gemm_bf16_wmma<false, false><<<gDD, 256, 0, stream>>>(hidbf, w1b, b1, nullptr, qf, MTOK_, DIM_, FFN_);
  gemm_bf16_wmma<false, false><<<gDD, 256, 0, stream>>>(hidbf, w2b, b2, nullptr, kf, MTOK_, DIM_, FFN_);

  // out = x2 + silu(g1)*g2
  swiglu_kernel<<<(MTOK_ * DIM_) / 256, 256, 0, stream>>>(x2, qf, kf, (float*)d_out, MTOK_ * DIM_);
}
